// ScaleDotProduct_4226247819312
// MI455X (gfx1250) — matrix operand order repaired and checked by reading, not compiled, not run on hardware
//
#include <hip/hip_runtime.h>

typedef __attribute__((ext_vector_type(16))) __bf16 v16bf;
typedef __attribute__((ext_vector_type(8)))  float  v8f;

#define S_LEN 2048
#define D_DIM 64
#define H_N   16
#define B_N   2
#define KT    32                 // kv tile per iteration
#define WAVES 4                  // waves per block, each owns 16 q rows
#define NTILE (S_LEN / KT)

// LDS generic pointer -> 32-bit LDS byte offset (shared aperture keeps the
// offset in addr[31:0]).
__device__ __forceinline__ unsigned lds_off(const void* p) {
    return (unsigned)(unsigned long long)p;
}

// Per-lane async DMA of one KT x D_DIM fp32 tile (8 KB) into LDS staging.
// 512 x 16B chunks / 128 threads = 4 global_load_async_to_lds_b128 per thread.
__device__ __forceinline__ void async_stage_tile(const float* gbase,
                                                 unsigned gbyte_off,
                                                 unsigned lds_base, int tid) {
    #pragma unroll
    for (int t = 0; t < 4; ++t) {
        const unsigned chunk = (unsigned)tid + 128u * t;
        const unsigned voff  = gbyte_off + chunk * 16u;
        const unsigned laddr = lds_base + chunk * 16u;
        asm volatile("global_load_async_to_lds_b128 %0, %1, %2"
                     :: "v"(laddr), "v"(voff), "s"(gbase)
                     : "memory");
    }
}

__global__ __launch_bounds__(128)
void fattn_bf16_wmma(const float* __restrict__ Q,
                     const float* __restrict__ K,
                     const float* __restrict__ V,
                     const unsigned char* __restrict__ mask,
                     float* __restrict__ out)
{
    __shared__ float  stageK[2][KT * D_DIM];   // async fp32 staging, double buffered
    __shared__ float  stageV[2][KT * D_DIM];
    __shared__ __bf16 Kb[KT * D_DIM];          // K tile bf16, row-major [kv][d]
    __shared__ __bf16 Vt[D_DIM * KT];          // V tile bf16, transposed [d][kv]
    __shared__ float  ldsP[WAVES * 16 * KT];   // per-wave P tile staging

    const int tid  = threadIdx.x;
    const int lane = tid & 31;
    const int wave = tid >> 5;
    const int l15  = lane & 15;
    const int hi   = lane >> 4;                // half-wave select (0/1)

    const int nQB = S_LEN / (16 * WAVES);      // 32 q-blocks per head
    const int qb  = blockIdx.x % nQB;
    const int h   = (blockIdx.x / nQB) % H_N;
    const int b   = blockIdx.x / (nQB * H_N);
    const int qBase = qb * (16 * WAVES) + wave * 16;

    const size_t headOff = ((size_t)(b * H_N + h)) * S_LEN * D_DIM;
    const float* Qh = Q + headOff;
    const float* Kh = K + headOff;
    const float* Vh = V + headOff;
    const unsigned char* Mb = mask + (size_t)b * S_LEN * S_LEN;

    const float NEG_INF = -__builtin_inff();
    const v8f vzero = {};

    // ---- Q A-fragments: 16x32 bf16, two K-chunks covering D=64 ----
    // A layout: lanes0-15 row M=lane hold K {0..7,16..23}; lanes16-31 {8..15,24..31}
    v16bf aq[2];
    {
        const float* qrow = Qh + (size_t)(qBase + l15) * D_DIM;
        #pragma unroll
        for (int c = 0; c < 2; ++c) {
            const int base = c * 32 + hi * 8;
            #pragma unroll
            for (int e = 0; e < 8; ++e) {
                aq[c][e]     = (__bf16)qrow[base + e];
                aq[c][e + 8] = (__bf16)qrow[base + 16 + e];
            }
        }
    }

    float m_i[8], l_i[8];
    v8f   O[4];
    #pragma unroll
    for (int r = 0; r < 8; ++r) { m_i[r] = NEG_INF; l_i[r] = 0.f; }
    #pragma unroll
    for (int n = 0; n < 4; ++n) O[n] = vzero;

    float* myP = ldsP + wave * (16 * KT);

    // ---- prologue: async-stage tile 0 ----
    async_stage_tile(Kh, 0u, lds_off(&stageK[0][0]), tid);
    async_stage_tile(Vh, 0u, lds_off(&stageV[0][0]), tid);

    for (int jt = 0; jt < NTILE; ++jt) {
        const int cur = jt & 1;
        const int j   = jt * KT;
        const bool has_next = (jt + 1) < NTILE;

        // prefetch next tile into the other buffer while we compute this one
        if (has_next) {
            const unsigned gb = (unsigned)(jt + 1) * (KT * D_DIM * 4u);
            async_stage_tile(Kh, gb, lds_off(&stageK[cur ^ 1][0]), tid);
            async_stage_tile(Vh, gb, lds_off(&stageV[cur ^ 1][0]), tid);
            asm volatile("s_wait_asynccnt 0x8" ::: "memory"); // current tile landed
        } else {
            asm volatile("s_wait_asynccnt 0x0" ::: "memory"); // drain
        }
        __syncthreads();   // staged data visible; prev-iter Kb/Vt/P reads done

        // ---- cooperative fp32 -> bf16 conversion passes (once per block) ----
        {
            const int kv    = tid >> 2;          // 0..31
            const int dbase = (tid & 3) * 16;    // 0,16,32,48
            const float* srcK = &stageK[cur][kv * D_DIM + dbase];
            const float* srcV = &stageV[cur][kv * D_DIM + dbase];
            #pragma unroll
            for (int q = 0; q < 16; ++q) {
                Kb[kv * D_DIM + dbase + q] = (__bf16)srcK[q];  // row-major
                Vt[(dbase + q) * KT + kv]  = (__bf16)srcV[q];  // transposed
            }
        }

        // ---- scores: S_tile[16 x 32] via 2 N-halves x 2 K-chunks of WMMA ----
        // (Kb read below is same-wave-produced? No: produced block-wide, so the
        //  mid-loop barrier below guards PV inputs; score inputs need Kb now.)
        __syncthreads();   // Kb/Vt ready for all waves

        v8f acc[2];
        #pragma unroll
        for (int hN = 0; hN < 2; ++hN) {
            acc[hN] = vzero;
            #pragma unroll
            for (int c = 0; c < 2; ++c) {
                // B layout 32x16: lanes0-15 col N=lane K {0..7,16..23}; lanes16-31 K {8..15,24..31}
                v16bf bk;
                const __bf16* krow = Kb + (hN * 16 + l15) * D_DIM + c * 32 + hi * 8;
                #pragma unroll
                for (int e = 0; e < 16; ++e) bk[e] = krow[e < 8 ? e : e + 8];
                acc[hN] = __builtin_amdgcn_wmma_f32_16x16x32_bf16(
                    false, aq[c], false, bk, (short)0, acc[hN], false, false);
            }
        }

        // ---- scale + mask (C layout: reg r -> row M=r+8*hi, col N=l15) ----
        float s[2][8];
        #pragma unroll
        for (int hN = 0; hN < 2; ++hN) {
            #pragma unroll
            for (int r = 0; r < 8; ++r) {
                const float v  = acc[hN][r] * 0.125f;   // 1/sqrt(64)
                const int   qr = qBase + r + 8 * hi;
                const unsigned char mb =
                    Mb[(size_t)qr * S_LEN + (j + hN * 16 + l15)];
                s[hN][r] = mb ? NEG_INF : v;
            }
        }

        // ---- online softmax (row reductions via 16-lane shfl_xor) ----
        float alpha[8];
        #pragma unroll
        for (int r = 0; r < 8; ++r) {
            float mr = fmaxf(s[0][r], s[1][r]);
            #pragma unroll
            for (int x = 1; x <= 8; x <<= 1)
                mr = fmaxf(mr, __shfl_xor(mr, x, 32));
            const float mnew = fmaxf(m_i[r], mr);
            alpha[r] = (m_i[r] == NEG_INF) ? 0.f : __expf(m_i[r] - mnew);
            float p0 = (mnew == NEG_INF) ? 0.f : __expf(s[0][r] - mnew);
            float p1 = (mnew == NEG_INF) ? 0.f : __expf(s[1][r] - mnew);
            s[0][r] = p0; s[1][r] = p1;
            float rs = p0 + p1;
            #pragma unroll
            for (int x = 1; x <= 8; x <<= 1)
                rs += __shfl_xor(rs, x, 32);
            l_i[r] = l_i[r] * alpha[r] + rs;
            m_i[r] = mnew;
        }
        #pragma unroll
        for (int n = 0; n < 4; ++n)
            #pragma unroll
            for (int r = 0; r < 8; ++r)
                O[n][r] *= alpha[r];

        // ---- P: C layout -> LDS -> A layout bf16 (per-wave region) ----
        #pragma unroll
        for (int hN = 0; hN < 2; ++hN)
            #pragma unroll
            for (int r = 0; r < 8; ++r)
                myP[(r + 8 * hi) * KT + hN * 16 + l15] = s[hN][r];

        v16bf ap;
        {
            const int base = hi * 8;
            #pragma unroll
            for (int e = 0; e < 8; ++e) {
                ap[e]     = (__bf16)myP[l15 * KT + base + e];
                ap[e + 8] = (__bf16)myP[l15 * KT + base + 16 + e];
            }
        }

        // ---- PV: O[16 x 64] += P(16x32) x V(32x64), 4 d-chunks ----
        #pragma unroll
        for (int n = 0; n < 4; ++n) {
            v16bf bv;
            const __bf16* col = Vt + (n * 16 + l15) * KT + hi * 8;
            #pragma unroll
            for (int e = 0; e < 16; ++e) bv[e] = col[e < 8 ? e : e + 8];
            O[n] = __builtin_amdgcn_wmma_f32_16x16x32_bf16(
                false, ap, false, bv, (short)0, O[n], false, false);
        }
    }

    // ---- epilogue: normalize (fully-masked rows -> 0, like nan_to_num) ----
    #pragma unroll
    for (int r = 0; r < 8; ++r) {
        const float inv = (l_i[r] > 0.f) ? (1.f / l_i[r]) : 0.f;
        const int   qr  = qBase + r + 8 * hi;
        float* orow = out + headOff + (size_t)qr * D_DIM;
        #pragma unroll
        for (int n = 0; n < 4; ++n)
            orow[n * 16 + l15] = O[n][r] * inv;
    }
}

extern "C" void kernel_launch(void* const* d_in, const int* in_sizes, int n_in,
                              void* d_out, int out_size, void* d_ws, size_t ws_size,
                              hipStream_t stream) {
    const float* Q = (const float*)d_in[0];
    const float* K = (const float*)d_in[1];
    const float* V = (const float*)d_in[2];
    const unsigned char* mask = (const unsigned char*)d_in[3];
    float* out = (float*)d_out;

    dim3 grid(B_N * H_N * (S_LEN / (16 * WAVES)));  // 2*16*32 = 1024
    dim3 block(128);
    hipLaunchKernelGGL(fattn_bf16_wmma, grid, block, 0, stream, Q, K, V, mask, out);
}
